// SimpleScanRNN_30674656428383
// MI455X (gfx1250) — matrix operand order repaired and checked by reading, not compiled, not run on hardware
//
#include <hip/hip_runtime.h>

// ---------------------------------------------------------------------------
// SimpleScanRNN for MI455X (gfx1250, wave32, WMMA)
//   B=64, T=1024, D=F=512
//   phase 1: d_out = x @ w_ih + bias          (parallel f16-WMMA GEMM)
//   phase 2: d_out[.,t,.] = tanh(d_out[.,t,.] + h @ w_hh), h updated   (scan)
// ---------------------------------------------------------------------------

typedef __attribute__((ext_vector_type(16))) _Float16 v16h;
typedef __attribute__((ext_vector_type(8)))  _Float16 v8h;
typedef __attribute__((ext_vector_type(8)))  float    v8f;

#define BATCH 64
#define SEQ   1024
#define DIM   512          // INPUT_DIM == FEATURES
#define KT_N  16           // 512 / 32 k-tiles per fragment row
#define NWG   16           // persistent workgroups in the scan kernel

// workspace layout (in _Float16 elements unless noted)
//   [0      , 262144) packed w_ih  (512 KB)
//   [262144 , 524288) packed w_hh  (512 KB)
//   [524288 , 589824) h ping-pong, 2 x (64*512) halves (128 KB)
//   byte 1179648     : grid-sync counter (u32)
#define WIH_OFF 0
#define WHH_OFF 262144
#define H_OFF   524288
#define H_ELEMS (BATCH * DIM)
#define CNT_BYTE_OFF 1179648

// ---------------------------------------------------------------------------
// Fragment layouts (CDNA5 ISA 7.12.2, wave32):
//  A (16x32 f16): lane L holds row m=L%16; half e -> k = kt*32 + (L/16)*8
//                 + (e<8 ? e : 16+(e-8))
//  B (32x16 f16): lane L holds col n=L%16; half e -> k = kt*32 + (L/16)*8 + (e<8 ? e : 16+(e-8))
//  C/D (16x16 f32): VGPR i, lane L -> m = i + 8*(L/16), n = L%16
// w_ih / w_hh are row-major [K=512][N=512]; we pre-pack B fragments so the
// inner loops load 32 contiguous bytes per lane (2 x global_load_b128).
// ---------------------------------------------------------------------------

__global__ __launch_bounds__(256)
void rnn_pack(const float* __restrict__ w_ih, const float* __restrict__ w_hh,
              _Float16* __restrict__ wih_p, _Float16* __restrict__ whh_p,
              _Float16* __restrict__ hbuf, unsigned* __restrict__ cnt)
{
    unsigned idx = blockIdx.x * 256u + threadIdx.x;
    if (idx < 524288u) {
        const bool hh = (idx & 262144u) != 0u;
        const float* __restrict__ W = hh ? w_hh : w_ih;
        _Float16* __restrict__ P    = hh ? whh_p : wih_p;
        unsigned r    = idx & 262143u;        // element within packed matrix
        unsigned e    = r & 15u;              // half index within lane
        unsigned lane = (r >> 4) & 31u;
        unsigned frag = r >> 9;               // (nt*KT_N + kt)
        unsigned nt   = frag >> 4;
        unsigned kt   = frag & 15u;
        unsigned k    = kt * 32u + (lane >> 4) * 8u + (e < 8u ? e : 8u + e);   // B-layout k
        unsigned n    = nt * 16u + (lane & 15u);
        P[r] = (_Float16)W[k * 512u + n];
    } else if (idx < 524288u + 2u * H_ELEMS) {
        hbuf[idx - 524288u] = (_Float16)0.0f;               // h0 = 0 (both bufs)
    } else if (idx == 524288u + 2u * H_ELEMS) {
        *cnt = 0u;                                          // grid-sync counter
    }
}

// ---------------------------------------------------------------------------
// Phase 1: x[BT x 512] @ w_ih[512 x 512] + bias -> d_out (row-major == [B,T,F])
// one 16x16 tile per wave; WG = 2 m-tiles x 4 n-tiles (8 waves).
// ---------------------------------------------------------------------------
__global__ __launch_bounds__(256)
void rnn_xproj_gemm(const float* __restrict__ x,
                    const _Float16* __restrict__ wih_p,
                    const float* __restrict__ bias,
                    float* __restrict__ out)
{
    const int wave = threadIdx.x >> 5;
    const int lane = threadIdx.x & 31;
    const int nblk = blockIdx.x & 7;          // 8 blocks of 64 cols
    const int mblk = blockIdx.x >> 3;         // 2048 blocks of 32 rows
    const int mt   = mblk * 2 + (wave & 1);   // 16-row tile, [0,4096)
    const int nt   = nblk * 4 + (wave >> 1);  // 16-col tile, [0,32)
    const int f    = nt * 16 + (lane & 15);
    const int arow = mt * 16 + (lane & 15);   // A row this lane feeds
    const int g8   = (lane >> 4) * 8;

    const float bf = bias[f];
    v8f acc;
#pragma unroll
    for (int i = 0; i < 8; ++i) acc[i] = bf;

    const float* __restrict__ rowp = x + (size_t)arow * DIM;
#pragma unroll 4
    for (int kt = 0; kt < KT_N; ++kt) {
        const int bk = kt * 32 + g8;
        float4 f0 = *(const float4*)(rowp + bk);
        float4 f1 = *(const float4*)(rowp + bk + 4);
        float4 f2 = *(const float4*)(rowp + bk + 16);
        float4 f3 = *(const float4*)(rowp + bk + 20);
        v16h a;
        a[0]  = (_Float16)f0.x; a[1]  = (_Float16)f0.y;
        a[2]  = (_Float16)f0.z; a[3]  = (_Float16)f0.w;
        a[4]  = (_Float16)f1.x; a[5]  = (_Float16)f1.y;
        a[6]  = (_Float16)f1.z; a[7]  = (_Float16)f1.w;
        a[8]  = (_Float16)f2.x; a[9]  = (_Float16)f2.y;
        a[10] = (_Float16)f2.z; a[11] = (_Float16)f2.w;
        a[12] = (_Float16)f3.x; a[13] = (_Float16)f3.y;
        a[14] = (_Float16)f3.z; a[15] = (_Float16)f3.w;
        v16h b = *(const v16h*)(wih_p + ((size_t)(nt * KT_N + kt) * 32 + lane) * 16);
        acc = __builtin_amdgcn_wmma_f32_16x16x32_f16(
                  false, a, false, b, (short)0, acc, false, false);
    }
    const int r0 = mt * 16 + (lane >> 4) * 8;   // first C row this lane owns
#pragma unroll
    for (int i = 0; i < 8; ++i)
        out[(size_t)(r0 + i) * DIM + f] = acc[i];
}

// ---------------------------------------------------------------------------
// Phase 2: persistent scan. 16 WGs x 8 waves; wave owns h-tile (mt, ntg).
// w_hh slice kept in VGPRs (16 x v16h = 128 VGPRs) for all 1024 steps.
// Ping-pong f16 h buffers + monotonic atomic counter for grid sync.
// ---------------------------------------------------------------------------
__global__ __launch_bounds__(256)
void rnn_scan(const _Float16* __restrict__ whh_p,
              _Float16* __restrict__ hbuf,
              unsigned* __restrict__ cnt,
              float* __restrict__ out)
{
    const int wave = threadIdx.x >> 5;
    const int lane = threadIdx.x & 31;
    const int mt   = wave & 3;                        // batch 16-row tile
    const int ntg  = blockIdx.x * 2 + (wave >> 2);    // feature 16-col tile [0,32)
    const int f    = ntg * 16 + (lane & 15);
    const int m    = mt * 16 + (lane & 15);           // A row (batch idx)
    const int g8   = (lane >> 4) * 8;
    const int b0   = mt * 16 + (lane >> 4) * 8;       // first C row (batch idx)

    // pin this wave's w_hh slice in registers: read once, reuse 1024 times
    v16h B[KT_N];
#pragma unroll
    for (int kt = 0; kt < KT_N; ++kt)
        B[kt] = *(const v16h*)(whh_p + ((size_t)(ntg * KT_N + kt) * 32 + lane) * 16);

    for (int t = 0; t < SEQ; ++t) {
        if (t) {  // wait until all 16 WGs finished step t-1
            if (threadIdx.x == 0) {
                while (__hip_atomic_load(cnt, __ATOMIC_ACQUIRE,
                                         __HIP_MEMORY_SCOPE_AGENT)
                       < (unsigned)(NWG * t))
                    __builtin_amdgcn_s_sleep(1);
            }
            __syncthreads();
        }
        const _Float16* __restrict__ hr = hbuf + (t & 1) * H_ELEMS;
        _Float16* __restrict__ hw       = hbuf + ((t + 1) & 1) * H_ELEMS;

        v8f acc;
#pragma unroll
        for (int i = 0; i < 8; ++i) acc[i] = 0.0f;

#pragma unroll
        for (int kt = 0; kt < KT_N; ++kt) {
            const int bk = kt * 32 + g8;
            v8h lo = *(const v8h*)(hr + (size_t)m * DIM + bk);
            v8h hi = *(const v8h*)(hr + (size_t)m * DIM + bk + 16);
            v16h a;
#pragma unroll
            for (int j = 0; j < 8; ++j) { a[j] = lo[j]; a[8 + j] = hi[j]; }
            acc = __builtin_amdgcn_wmma_f32_16x16x32_f16(
                      false, a, false, B[kt], (short)0, acc, false, false);
        }

#pragma unroll
        for (int i = 0; i < 8; ++i) {
            const size_t oidx = ((size_t)(b0 + i) * SEQ + t) * DIM + f;
            const float v = tanhf(out[oidx] + acc[i]);   // xp + h@w_hh
            out[oidx] = v;
            hw[(size_t)(b0 + i) * DIM + f] = (_Float16)v;
        }

        __syncthreads();                   // all waves in WG done writing
        if (threadIdx.x == 0) {
            __threadfence();               // make h/out visible device-wide
            atomicAdd(cnt, 1u);
        }
    }
}

// ---------------------------------------------------------------------------
extern "C" void kernel_launch(void* const* d_in, const int* in_sizes, int n_in,
                              void* d_out, int out_size, void* d_ws, size_t ws_size,
                              hipStream_t stream)
{
    (void)in_sizes; (void)n_in; (void)out_size; (void)ws_size;
    const float* x    = (const float*)d_in[0];   // [64,1024,512]
    const float* w_ih = (const float*)d_in[1];   // [512,512]
    const float* w_hh = (const float*)d_in[2];   // [512,512]
    const float* bias = (const float*)d_in[3];   // [512]
    float* out = (float*)d_out;                  // [64,1024,512]

    _Float16* wsH   = (_Float16*)d_ws;
    _Float16* wih_p = wsH + WIH_OFF;
    _Float16* whh_p = wsH + WHH_OFF;
    _Float16* hbuf  = wsH + H_OFF;
    unsigned* cnt   = (unsigned*)((char*)d_ws + CNT_BYTE_OFF);

    // 1) pack weights to f16 B-fragments, zero h, reset counter
    {
        const unsigned total = 524288u + 2u * H_ELEMS + 1u;
        rnn_pack<<<(total + 255u) / 256u, 256, 0, stream>>>(
            w_ih, w_hh, wih_p, whh_p, hbuf, cnt);
    }
    // 2) x_proj + bias -> d_out  (M=65536, N=512, K=512)
    rnn_xproj_gemm<<<16384, 256, 0, stream>>>(x, wih_p, bias, out);
    // 3) sequential scan, persistent grid
    rnn_scan<<<NWG, 256, 0, stream>>>(whh_p, hbuf, cnt, out);
}
